// DilatedConvModel_79542794322624
// MI455X (gfx1250) — matrix operand order repaired and checked by reading, not compiled, not run on hardware
//
#include <hip/hip_runtime.h>
#include <hip/hip_bf16.h>
#include <stdint.h>

// ---------------------------------------------------------------------------
// Sizes from the reference
#define B_    64
#define S_    8
#define T_    8192
#define T3_   8166            // 8192 - 2 - 6 - 18
#define T3P   8168            // row stride (bf16) padded so rows are 16B-aligned
#define NSEQ  (B_*S_)         // 512
#define TOUT  512             // final-layer timesteps produced per workgroup
#define NCHUNK ((T3_ + TOUT - 1)/TOUT)   // 16
#define EPSV  1e-8f

// LDS row pitch (bf16 elems): 24 => 48B rows, both uint4 halves 16B-aligned,
// 12-dword row stride gives conflict-free 4-bank windows per lane group.
#define PITCH 24

typedef __attribute__((ext_vector_type(16))) __bf16 v16bf;
typedef __attribute__((ext_vector_type(8)))  float  v8f;

union BF16Frag { v16bf v; __bf16 b[16]; uint4 q[2]; };
union FragC    { v8f   v; float f[8]; };
union Pack8    { __bf16 b[8]; uint4 q; };

// Native converts: compiler selects hw bf16 cvt (RNE), not 3-op emulation.
__device__ __forceinline__ __bf16 f2bf(float f) { return (__bf16)f; }
__device__ __forceinline__ float  bf2f(__bf16 h) { return (float)h; }

__device__ __forceinline__ v8f wmma_bf16(v16bf a, v16bf b, v8f c) {
  // D = A(16x32 bf16) * B(32x16 bf16) + C(f32)
  return __builtin_amdgcn_wmma_f32_16x16x32_bf16(false, a, false, b,
                                                 (short)0, c, false, false);
}

// A-matrix (16x32 bf16) half->K map: lanes 0-15 hold K=0..15,
// lanes 16-31 hold K=16..31
__device__ __forceinline__ int a_kidx(int h, int g) {
  return g * 16 + h;
}

// Load one 16-channel LDS row (32B) into a fragment's 16 halves
__device__ __forceinline__ void ld_row16(const __bf16* row, BF16Frag& f) {
  f.q[0] = *(const uint4*)(row);
  f.q[1] = *(const uint4*)(row + 8);
}

// ===========================================================================
// Fused stimulus stack: 1 -> 16 (d=1, VALU) -> 16 (d=3, WMMA) -> 16 (d=9, WMMA)
// grid: (NCHUNK, NSEQ), block: 256 (8 waves).  out: bf16 [NSEQ][16][T3P]
// LDS activations are [t][ch]; branch-free zero row feeds the k=2 K-slice
// for lanes 16-31.
// ===========================================================================
__global__ __launch_bounds__(256) void stim_stack_kernel(
    const float* __restrict__ stim,
    const float* __restrict__ w1, const float* __restrict__ b1,
    const float* __restrict__ w2, const float* __restrict__ b2,
    const float* __restrict__ w3, const float* __restrict__ b3,
    __bf16* __restrict__ out)
{
  const int chunk = blockIdx.x;
  const int seq   = blockIdx.y;           // seq = s*B + b  (transpose (1,0,3,2))
  const int t0    = chunk * TOUT;
  const int s     = seq / B_;
  const int b     = seq % B_;
  const float* x  = stim + ((size_t)(b * S_ + s)) * T_;   // stimulus[b][s][:,0]

  __shared__ float sin_[TOUT + 32];                        // need TOUT+26
  __shared__ __align__(16) __bf16 s1t[TOUT + 40][PITCH];   // rows 0..535 valid
  __shared__ __align__(16) __bf16 s2t[TOUT + 34][PITCH];   // rows written to 543
  const int ZR1 = TOUT + 39;   // 551: zero row (reads reach <= 549)
  const int ZR2 = TOUT + 33;   // 545: zero row (writes reach <= 543)

  const int tid  = threadIdx.x;
  const int lane = tid & 31;
  const int wave = tid >> 5;
  const int g    = lane >> 4;
  const int mn   = lane & 15;

  if (tid < PITCH) { s1t[ZR1][tid] = (__bf16)0.f; s2t[ZR2][tid] = (__bf16)0.f; }
  for (int i = tid; i < TOUT + 26; i += 256) {
    int t = t0 + i;
    sin_[i] = x[(t < T_) ? t : (T_ - 1)];   // clamp: garbage only hits masked outputs
  }
  __syncthreads();

  // layer 1: Cin=1, scalar VALU; one thread computes all 16 channels of a row
  for (int t = tid; t < TOUT + 24; t += 256) {
    float i0 = sin_[t], i1 = sin_[t + 1], i2 = sin_[t + 2];
    Pack8 lo, hi;
#pragma unroll
    for (int c = 0; c < 8; ++c) {
      float v0 = b1[c]     + w1[c*3+0]*i0 + w1[c*3+1]*i1 + w1[c*3+2]*i2;
      float v1 = b1[c + 8] + w1[(c+8)*3+0]*i0 + w1[(c+8)*3+1]*i1 + w1[(c+8)*3+2]*i2;
      lo.b[c] = f2bf(fmaxf(v0, 0.f));
      hi.b[c] = f2bf(fmaxf(v1, 0.f));
    }
    *(uint4*)&s1t[t][0] = lo.q;
    *(uint4*)&s1t[t][8] = hi.q;
  }

  // A fragments for layers 2/3: kappa = k*16 + ci, K padded 48->64
  BF16Frag a20, a21, a30, a31;
#pragma unroll
  for (int h = 0; h < 16; ++h) {
    int K = a_kidx(h, g);
    { int k = K >> 4, ci = K & 15;                          // slice 0: kappa in [0,32)
      a20.b[h] = f2bf(w2[(mn*16 + ci)*3 + k]);
      a30.b[h] = f2bf(w3[(mn*16 + ci)*3 + k]); }
    if (K < 16) {                                           // slice 1: kappa = 32+K < 48
      a21.b[h] = f2bf(w2[(mn*16 + K)*3 + 2]);
      a31.b[h] = f2bf(w3[(mn*16 + K)*3 + 2]);
    } else { a21.b[h] = (__bf16)0.f; a31.b[h] = (__bf16)0.f; }
  }
  __syncthreads();

  // layer 2: d=3; tiles cover TOUT+18 positions
  const int NT2 = (TOUT + 18 + 15) / 16;                    // 34
  for (int tt = wave; tt < NT2; tt += 8) {
    int tpos = tt * 16 + mn;
    BF16Frag bf0, bf1;
    ld_row16(&s1t[tpos + g * 3][0], bf0);                   // k=g, ci=0..15
    ld_row16(&s1t[g ? ZR1 : tpos + 6][0], bf1);             // k=2 (zero for g=1)
    v8f c = {};
    c = wmma_bf16(a20.v, bf0.v, c);
    c = wmma_bf16(a21.v, bf1.v, c);
    FragC fc; fc.v = c;
    Pack8 pk;
#pragma unroll
    for (int r = 0; r < 8; ++r)
      pk.b[r] = f2bf(fmaxf(fc.f[r] + b2[r + g * 8], 0.f));
    *(uint4*)&s2t[tpos][g * 8] = pk.q;
  }
  __syncthreads();

  // layer 3: d=9; write bf16 to global (stride T3P)
  const int NT3 = TOUT / 16;                                // 32
  for (int tt = wave; tt < NT3; tt += 8) {
    int tpos = tt * 16 + mn;
    BF16Frag bf0, bf1;
    ld_row16(&s2t[tpos + g * 9][0], bf0);
    ld_row16(&s2t[g ? ZR2 : tpos + 18][0], bf1);
    v8f c = {};
    c = wmma_bf16(a30.v, bf0.v, c);
    c = wmma_bf16(a31.v, bf1.v, c);
    FragC fc; fc.v = c;
    int t3 = t0 + tpos;
    if (t3 < T3_) {
#pragma unroll
      for (int r = 0; r < 8; ++r) {
        int m = r + g * 8;
        out[((size_t)seq * 16 + m) * T3P + t3] = f2bf(fmaxf(fc.f[r] + b3[m], 0.f));
      }
    }
  }
}

// ===========================================================================
// Fused EEG stack: 64 -> 8 (1x1, WMMA) -> 16 (d=1) -> 16 (d=3) -> 16 (d=9)
// grid: (NCHUNK, B), block 256.  out: bf16 [B][16][T3P]
// ===========================================================================
__global__ __launch_bounds__(256) void eeg_stack_kernel(
    const float* __restrict__ eeg,
    const float* __restrict__ w0, const float* __restrict__ b0v,
    const float* __restrict__ w1, const float* __restrict__ b1v,
    const float* __restrict__ w2, const float* __restrict__ b2v,
    const float* __restrict__ w3, const float* __restrict__ b3v,
    __bf16* __restrict__ out)
{
  const int chunk = blockIdx.x;
  const int bb    = blockIdx.y;
  const int t0    = chunk * TOUT;

  __shared__ __align__(16) __bf16 x0t[TOUT + 40][8];       // 1x1 out (no relu)
  __shared__ __align__(16) __bf16 x1t[TOUT + 40][PITCH];
  __shared__ __align__(16) __bf16 x2t[TOUT + 34][PITCH];
  const int ZR0 = TOUT + 39;   // 551 (reads <= 545)
  const int ZR1 = TOUT + 39;   // 551 (reads <= 549)
  const int ZR2 = TOUT + 33;   // 545 (writes <= 543)

  const int tid  = threadIdx.x;
  const int lane = tid & 31;
  const int wave = tid >> 5;
  const int g    = lane >> 4;
  const int mn   = lane & 15;

  if (tid < PITCH) { x1t[ZR1][tid] = (__bf16)0.f; x2t[ZR2][tid] = (__bf16)0.f; }
  if (tid < 8)     { x0t[ZR0][tid] = (__bf16)0.f; }

  // ---- A fragments -------------------------------------------------------
  // 1x1: A[m][ci], M padded 8->16, K=64 -> two slices
  BF16Frag ae0, ae1;
#pragma unroll
  for (int h = 0; h < 16; ++h) {
    int K = a_kidx(h, g);
    ae0.b[h] = (mn < 8) ? f2bf(w0[mn * 64 + K])      : (__bf16)0.f;
    ae1.b[h] = (mn < 8) ? f2bf(w0[mn * 64 + 32 + K]) : (__bf16)0.f;
  }
  // layer1: Cin=8, kappa = k*8 + ci, K=24 padded to 32 (one WMMA)
  BF16Frag a1;
#pragma unroll
  for (int h = 0; h < 16; ++h) {
    if (h < 8)       a1.b[h] = f2bf(w1[(mn*8 + h)*3 + g]);       // kappa = h + g*8
    else if (g == 0) a1.b[h] = f2bf(w1[(mn*8 + (h-8))*3 + 2]);   // kappa = 16+(h-8)
    else             a1.b[h] = (__bf16)0.f;                       // kappa >= 24
  }
  // layers 2/3: Cin=16
  BF16Frag a20, a21, a30, a31;
#pragma unroll
  for (int h = 0; h < 16; ++h) {
    int K = a_kidx(h, g);
    { int k = K >> 4, ci = K & 15;
      a20.b[h] = f2bf(w2[(mn*16 + ci)*3 + k]);
      a30.b[h] = f2bf(w3[(mn*16 + ci)*3 + k]); }
    if (K < 16) {
      a21.b[h] = f2bf(w2[(mn*16 + K)*3 + 2]);
      a31.b[h] = f2bf(w3[(mn*16 + K)*3 + 2]);
    } else { a21.b[h] = (__bf16)0.f; a31.b[h] = (__bf16)0.f; }
  }

  // ---- 1x1 conv (no relu): tiles cover TOUT+26 positions -----------------
  const int NT0 = (TOUT + 26 + 15) / 16;                    // 34
  for (int tt = wave; tt < NT0; tt += 8) {
    int tpos = tt * 16 + mn;
    int t    = t0 + tpos;
    int tc   = (t < T_) ? t : (T_ - 1);   // clamp: garbage only hits masked outputs
    const float* ep = eeg + ((size_t)bb * T_ + tc) * 64;
    const float4* e4  = (const float4*)(ep + g * 16);       // ci = g*16 + 0..15
    const float4* e4b = (const float4*)(ep + 32 + g * 16);  // ci = 32 + g*16 + 0..15
    float4 f0 = e4[0],  f1 = e4[1],  f2 = e4[2],  f3 = e4[3];
    float4 h0 = e4b[0], h1 = e4b[1], h2 = e4b[2], h3 = e4b[3];
    float lo[16] = { f0.x,f0.y,f0.z,f0.w, f1.x,f1.y,f1.z,f1.w,
                     f2.x,f2.y,f2.z,f2.w, f3.x,f3.y,f3.z,f3.w };
    float hi[16] = { h0.x,h0.y,h0.z,h0.w, h1.x,h1.y,h1.z,h1.w,
                     h2.x,h2.y,h2.z,h2.w, h3.x,h3.y,h3.z,h3.w };
    BF16Frag bf0, bf1;
#pragma unroll
    for (int h = 0; h < 16; ++h) { bf0.b[h] = f2bf(lo[h]); bf1.b[h] = f2bf(hi[h]); }
    v8f c = {};
    c = wmma_bf16(ae0.v, bf0.v, c);
    c = wmma_bf16(ae1.v, bf1.v, c);
    FragC fc; fc.v = c;
    if (g == 0) {                                           // only rows m<8 exist
      Pack8 pk;
#pragma unroll
      for (int r = 0; r < 8; ++r) pk.b[r] = f2bf(fc.f[r] + b0v[r]);
      *(uint4*)&x0t[tpos][0] = pk.q;
    }
  }
  __syncthreads();

  // ---- layer 1: 8->16, d=1 -----------------------------------------------
  // halves 0..7: k = g, ci=0..7 ; halves 8..15: (g==0) k=2 else zero
  const int NT1 = (TOUT + 24 + 15) / 16;                    // 34
  for (int tt = wave; tt < NT1; tt += 8) {
    int tpos = tt * 16 + mn;
    BF16Frag bf0;
    bf0.q[0] = *(const uint4*)&x0t[tpos + g][0];
    bf0.q[1] = *(const uint4*)&x0t[g ? ZR0 : tpos + 2][0];
    v8f c = {};
    c = wmma_bf16(a1.v, bf0.v, c);
    FragC fc; fc.v = c;
    Pack8 pk;
#pragma unroll
    for (int r = 0; r < 8; ++r)
      pk.b[r] = f2bf(fmaxf(fc.f[r] + b1v[r + g * 8], 0.f));
    *(uint4*)&x1t[tpos][g * 8] = pk.q;
  }
  __syncthreads();

  // ---- layer 2: 16->16, d=3 ----------------------------------------------
  const int NT2 = (TOUT + 18 + 15) / 16;                    // 34
  for (int tt = wave; tt < NT2; tt += 8) {
    int tpos = tt * 16 + mn;
    BF16Frag bf0, bf1;
    ld_row16(&x1t[tpos + g * 3][0], bf0);
    ld_row16(&x1t[g ? ZR1 : tpos + 6][0], bf1);
    v8f c = {};
    c = wmma_bf16(a20.v, bf0.v, c);
    c = wmma_bf16(a21.v, bf1.v, c);
    FragC fc; fc.v = c;
    Pack8 pk;
#pragma unroll
    for (int r = 0; r < 8; ++r)
      pk.b[r] = f2bf(fmaxf(fc.f[r] + b2v[r + g * 8], 0.f));
    *(uint4*)&x2t[tpos][g * 8] = pk.q;
  }
  __syncthreads();

  // ---- layer 3: 16->16, d=9, write global --------------------------------
  const int NT3 = TOUT / 16;
  for (int tt = wave; tt < NT3; tt += 8) {
    int tpos = tt * 16 + mn;
    BF16Frag bf0, bf1;
    ld_row16(&x2t[tpos + g * 9][0], bf0);
    ld_row16(&x2t[g ? ZR2 : tpos + 18][0], bf1);
    v8f c = {};
    c = wmma_bf16(a30.v, bf0.v, c);
    c = wmma_bf16(a31.v, bf1.v, c);
    FragC fc; fc.v = c;
    int t3 = t0 + tpos;
    if (t3 < T3_) {
#pragma unroll
      for (int r = 0; r < 8; ++r) {
        int m = r + g * 8;
        out[((size_t)bb * 16 + m) * T3P + t3] = f2bf(fmaxf(fc.f[r] + b3v[m], 0.f));
      }
    }
  }
}

// ===========================================================================
// Row L2 norms of bf16 [rows][T3P] -> dst[row] = max(||row(0:T3)||, EPS)
// ===========================================================================
__global__ __launch_bounds__(256) void rownorm_kernel(
    const __bf16* __restrict__ src, float* __restrict__ dst)
{
  const int row = blockIdx.x;
  const __bf16* p = src + (size_t)row * T3P;
  float acc = 0.f;
  const int NV = (T3_ / 8) * 8;                             // 8160
  for (int i = threadIdx.x * 8; i < NV; i += 256 * 8) {
    Pack8 q; q.q = *(const uint4*)(p + i);
#pragma unroll
    for (int k = 0; k < 8; ++k) { float v = bf2f(q.b[k]); acc += v * v; }
  }
  if (threadIdx.x < T3_ - NV) {                             // 6-element tail
    float v = bf2f(p[NV + threadIdx.x]);
    acc += v * v;
  }
  __shared__ float red[256];
  red[threadIdx.x] = acc;
  __syncthreads();
  for (int s = 128; s > 0; s >>= 1) {
    if (threadIdx.x < s) red[threadIdx.x] += red[threadIdx.x + s];
    __syncthreads();
  }
  if (threadIdx.x == 0) dst[row] = fmaxf(sqrtf(red[0]), EPSV);
}

// ===========================================================================
// dot[seq][i][j] = sum_t st[seq][i][t] * x[b][j][t], K = T3, via WMMA.
// grid NSEQ, block 128 (4 waves split K, LDS partial reduce).
// Fast path: A/B fragments are two aligned uint4 global loads each.
// ===========================================================================
__global__ __launch_bounds__(128) void dot_kernel(
    const __bf16* __restrict__ st, const __bf16* __restrict__ xe,
    float* __restrict__ dotb)
{
  const int seq  = blockIdx.x;
  const int b    = seq % B_;
  const int lane = threadIdx.x & 31;
  const int wave = threadIdx.x >> 5;
  const int g    = lane >> 4;
  const int mn   = lane & 15;
  const __bf16* ap = st + (size_t)seq * 16 * T3P + (size_t)mn * T3P;
  const __bf16* bp = xe + (size_t)b   * 16 * T3P + (size_t)mn * T3P;

  v8f c = {};
  const int KC = (T3_ + 31) / 32;                           // 256 K-chunks
  for (int kc = wave; kc < KC; kc += 4) {
    int tb = kc * 32;
    BF16Frag af, bf;
    if (tb + 32 <= T3_) {
      af.q[0] = *(const uint4*)(ap + tb + g * 16);          // K = g*16 + 0..7
      af.q[1] = *(const uint4*)(ap + tb + g * 16 + 8);      // K = g*16 + 8..15
      bf.q[0] = *(const uint4*)(bp + tb + g * 16);          // K = g*16 + 0..7
      bf.q[1] = *(const uint4*)(bp + tb + g * 16 + 8);      // K = g*16 + 8..15
    } else {
#pragma unroll
      for (int h = 0; h < 16; ++h) {
        int ta = tb + a_kidx(h, g);
        af.b[h] = (ta < T3_) ? ap[ta] : (__bf16)0.f;
        int tx = tb + g * 16 + h;
        bf.b[h] = (tx < T3_) ? bp[tx] : (__bf16)0.f;
      }
    }
    c = wmma_bf16(af.v, bf.v, c);
  }

  __shared__ float part[4][16][16];
  FragC fc; fc.v = c;
#pragma unroll
  for (int r = 0; r < 8; ++r) part[wave][r + g * 8][mn] = fc.f[r];
  __syncthreads();
  for (int e = threadIdx.x; e < 256; e += 128) {
    int i = e >> 4, j = e & 15;
    dotb[(size_t)seq * 256 + e] =
        part[0][i][j] + part[1][i][j] + part[2][i][j] + part[3][i][j];
  }
}

// ===========================================================================
// Head: out[b*S+s] = b_lin + sum_ij w_lin[i*16+j] * dot/(n_st*n_x)
// ===========================================================================
__global__ __launch_bounds__(256) void head_kernel(
    const float* __restrict__ dotb, const float* __restrict__ nst,
    const float* __restrict__ nx, const float* __restrict__ wlin,
    const float* __restrict__ blin, float* __restrict__ outp)
{
  int tid = blockIdx.x * 256 + threadIdx.x;
  if (tid >= B_ * S_) return;
  int b = tid / S_, s = tid % S_;
  int seq = s * B_ + b;
  float inv_nx[16];
#pragma unroll
  for (int j = 0; j < 16; ++j) inv_nx[j] = 1.f / nx[b * 16 + j];
  float acc = blin[0];
  for (int i = 0; i < 16; ++i) {
    float inv_i = 1.f / nst[seq * 16 + i];
#pragma unroll
    for (int j = 0; j < 16; ++j)
      acc += wlin[i * 16 + j] * dotb[(size_t)seq * 256 + i * 16 + j] * inv_i * inv_nx[j];
  }
  outp[tid] = acc;
}

// ===========================================================================
extern "C" void kernel_launch(void* const* d_in, const int* in_sizes, int n_in,
                              void* d_out, int out_size, void* d_ws, size_t ws_size,
                              hipStream_t stream)
{
  const float* eeg   = (const float*)d_in[0];
  const float* stim  = (const float*)d_in[1];
  const float* w_eeg = (const float*)d_in[2];
  const float* b_eeg = (const float*)d_in[3];
  const float* w_e1  = (const float*)d_in[4];  const float* b_e1 = (const float*)d_in[5];
  const float* w_e2  = (const float*)d_in[6];  const float* b_e2 = (const float*)d_in[7];
  const float* w_e3  = (const float*)d_in[8];  const float* b_e3 = (const float*)d_in[9];
  const float* w_s1  = (const float*)d_in[10]; const float* b_s1 = (const float*)d_in[11];
  const float* w_s2  = (const float*)d_in[12]; const float* b_s2 = (const float*)d_in[13];
  const float* w_s3  = (const float*)d_in[14]; const float* b_s3 = (const float*)d_in[15];
  const float* w_lin = (const float*)d_in[16]; const float* b_lin= (const float*)d_in[17];

  char* ws = (char*)d_ws;
  size_t off = 0;
  __bf16* s3 = (__bf16*)(ws + off); off += (size_t)NSEQ * 16 * T3P * 2;
  __bf16* x3 = (__bf16*)(ws + off); off += (size_t)B_   * 16 * T3P * 2;
  float* dotb = (float*)(ws + off); off += (size_t)NSEQ * 256 * 4;
  float* nst  = (float*)(ws + off); off += (size_t)NSEQ * 16 * 4;
  float* nx   = (float*)(ws + off); off += (size_t)B_   * 16 * 4;

  stim_stack_kernel<<<dim3(NCHUNK, NSEQ), 256, 0, stream>>>(
      stim, w_s1, b_s1, w_s2, b_s2, w_s3, b_s3, s3);
  eeg_stack_kernel<<<dim3(NCHUNK, B_), 256, 0, stream>>>(
      eeg, w_eeg, b_eeg, w_e1, b_e1, w_e2, b_e2, w_e3, b_e3, x3);
  rownorm_kernel<<<NSEQ * 16, 256, 0, stream>>>(s3, nst);
  rownorm_kernel<<<B_ * 16, 256, 0, stream>>>(x3, nx);
  dot_kernel<<<NSEQ, 128, 0, stream>>>(s3, x3, dotb);
  head_kernel<<<2, 256, 0, stream>>>(dotb, nst, nx, w_lin, b_lin, (float*)d_out);
}
